// SelfAttention2D_40750649705061
// MI455X (gfx1250) — matrix operand order repaired and checked by reading, not compiled, not run on hardware
//
#include <hip/hip_runtime.h>
#include <hip/hip_bf16.h>

// ---------------------------------------------------------------------------
// SelfAttention2D for MI455X (gfx1250): GroupNorm -> QKV -> flash-attention
// -> proj + residual, all GEMMs via v_wmma_f32_16x16x32_f16 (wave32 WMMA).
// Round 4: fully-unrolled K loops (immediate-offset addressing, no per-iter
// address VALU -> no WAR v_nops), attention key loop unrolled x2.
// ---------------------------------------------------------------------------

typedef __attribute__((ext_vector_type(16))) _Float16     v16h;
typedef __attribute__((ext_vector_type(8)))  float        v8f;
typedef __attribute__((ext_vector_type(8)))  unsigned int v8u;

#define NB   4
#define CCH  256
#define NPIX 4096
#define NHD  4
#define DH   64
// 1/sqrt(dh) * log2(e): attention scores computed in base-2 domain.
#define QSCALE 0.1803368801111204f

__device__ __forceinline__ v8f wmma_f16(v16h a, v16h b, v8f c) {
  return __builtin_amdgcn_wmma_f32_16x16x32_f16(false, a, false, b, (short)0, c,
                                                false, false);
}

// A-fragment (16x32 f16, M x K): row-major source, K contiguous.
// lane m = L&15; K base 0 (lanes 0-15) / 8 (lanes 16-31);
// VGPR r holds f16 pair K = kb + 2*(r&3) + 16*(r>>2).
__device__ __forceinline__ v16h load_a(const _Float16* base, int row0, int ld,
                                       int koff, int lane) {
  const int m  = lane & 15;
  const int kb = (lane >> 4) << 3;
  const _Float16* p = base + (size_t)(row0 + m) * ld + koff + kb;
  v8u u;
#pragma unroll
  for (int r = 0; r < 8; ++r) {
    const int k = ((r & 3) << 1) + ((r >> 2) << 4);
    u[r] = *(const unsigned int*)(p + k);
  }
  return __builtin_bit_cast(v16h, u);
}

// B-fragment (32x16 f16, K x N): source is B^T row-major [Ncols x K], K contig.
// lane n = L&15; K base 0 (lanes 0-15) / 8 (lanes 16-31);
// VGPR r holds f16 pair K = kb + 2*(r&3) + 16*(r>>2).
__device__ __forceinline__ v16h load_b(const _Float16* bt, int col0, int ld,
                                       int koff, int lane) {
  const int n  = lane & 15;
  const int kb = (lane >> 4) << 3;
  const _Float16* p = bt + (size_t)(col0 + n) * ld + koff + kb;
  v8u u;
#pragma unroll
  for (int r = 0; r < 8; ++r)
    u[r] = *(const unsigned int*)(p + ((r & 3) << 1) + ((r >> 2) << 4));
  return __builtin_bit_cast(v16h, u);
}

// ---------------------------------------------------------------------------
// Kernel 0: f32 -> f16 conversion (weights)
// ---------------------------------------------------------------------------
__global__ void cvt_f32_f16(const float* __restrict__ a, _Float16* __restrict__ o,
                            int n) {
  int i = blockIdx.x * blockDim.x + threadIdx.x;
  if (i < n) o[i] = (_Float16)a[i];
}

// ---------------------------------------------------------------------------
// Kernel 1: GroupNorm(8, C).  One block per (b, g); group region is contiguous
// (32 channels x 4096 pixels).  Writes h as f16 in [B, N, C].
// ---------------------------------------------------------------------------
__global__ __launch_bounds__(256) void groupnorm_kernel(
    const float* __restrict__ x, const float* __restrict__ gw,
    const float* __restrict__ gb, _Float16* __restrict__ h) {
  const int bg = blockIdx.x;         // 0..31
  const int b = bg >> 3, g = bg & 7;
  const int lane = threadIdx.x & 31, wave = threadIdx.x >> 5;
  const float* xb = x + ((size_t)b * CCH + g * 32) * NPIX;
  const int GElems = 32 * NPIX;      // 131072

  float s = 0.f, ss = 0.f;
  for (int i = threadIdx.x; i < GElems; i += 256) {
    float v = xb[i];
    s += v;
    ss += v * v;
  }
#pragma unroll
  for (int mask = 1; mask < 32; mask <<= 1) {
    s  += __shfl_xor(s, mask, 32);
    ss += __shfl_xor(ss, mask, 32);
  }
  __shared__ float ls[8], lss[8];
  if (lane == 0) { ls[wave] = s; lss[wave] = ss; }
  __syncthreads();
  float ts = 0.f, tss = 0.f;
#pragma unroll
  for (int w = 0; w < 8; ++w) { ts += ls[w]; tss += lss[w]; }
  const float inv_n = 1.0f / (float)GElems;
  const float mean = ts * inv_n;
  const float var = tss * inv_n - mean * mean;
  const float rstd = rsqrtf(var + 1e-5f);

  for (int i = threadIdx.x; i < GElems; i += 256) {
    const int cg = i >> 12;
    const int n = i & (NPIX - 1);
    const int c = g * 32 + cg;
    float v = (xb[i] - mean) * rstd * gw[c] + gb[c];
    h[((size_t)b * NPIX + n) * CCH + c] = (_Float16)v;
  }
}

// ---------------------------------------------------------------------------
// Kernel 2: QKV GEMM.  One wave -> 32x64 tile (8 WMMA accs, shared B frags).
// K loop fully unrolled: all loads are base + immediate offset, fragments get
// distinct SSA registers per step, scheduler hoists loads across steps.
// ---------------------------------------------------------------------------
__global__ __launch_bounds__(128) void qkv_gemm_kernel(
    const _Float16* __restrict__ w,    // [768, 256] row-major
    const _Float16* __restrict__ h,    // [B, N, C]
    const float* __restrict__ bias,    // [768]
    _Float16* __restrict__ qo,         // [B*NH, N, DH]
    _Float16* __restrict__ ko,         // [B*NH, N, DH]
    _Float16* __restrict__ vo) {       // [B*NH, DH, N]
  const int lane = threadIdx.x & 31, wave = threadIdx.x >> 5;
  const int b = blockIdx.z;
  const int n0 = blockIdx.x * 64;
  const int m0 = blockIdx.y * 128 + wave * 32;
  const _Float16* bt = h + (size_t)b * NPIX * CCH;

  v8f acc0[4] = {}, acc1[4] = {};
#pragma unroll
  for (int k = 0; k < CCH; k += 32) {
    v16h a0 = load_a(w, m0, CCH, k, lane);
    v16h a1 = load_a(w, m0 + 16, CCH, k, lane);
    v16h bf[4];
#pragma unroll
    for (int t = 0; t < 4; ++t) bf[t] = load_b(bt, n0 + t * 16, CCH, k, lane);
#pragma unroll
    for (int t = 0; t < 4; ++t) {
      acc0[t] = wmma_f16(a0, bf[t], acc0[t]);
      acc1[t] = wmma_f16(a1, bf[t], acc1[t]);
    }
  }

  const int col = lane & 15;
  const int hi8 = (lane >> 4) << 3;
#pragma unroll
  for (int ms = 0; ms < 2; ++ms) {
    const int m0s = m0 + ms * 16;
    const int third = m0s >> 8;          // 0=q 1=k 2=v (uniform per wave)
    const int head = (m0s & 255) >> 6;
    const size_t bh = (size_t)(b * NHD + head);
    v8f* acc = ms ? acc1 : acc0;
#pragma unroll
    for (int t = 0; t < 4; ++t) {
      const int n = n0 + t * 16 + col;
#pragma unroll
      for (int v = 0; v < 8; ++v) {
        const int rl = v + hi8;
        const int o = m0s + rl;
        const int d = (m0s & 63) + rl;
        float val = acc[t][v] + bias[o];
        if (third == 0)
          qo[(bh * NPIX + n) * DH + d] = (_Float16)(val * QSCALE);
        else if (third == 1)
          ko[(bh * NPIX + n) * DH + d] = (_Float16)val;
        else
          vo[(bh * DH + d) * NPIX + n] = (_Float16)val;
      }
    }
  }
}

// ---------------------------------------------------------------------------
// Kernel 3: flash attention.  One wave per 16 query rows; sweep 4096 keys in
// chunks of 32 (unrolled x2 so two chunks' fragments interleave).  All 8
// fragment loads issued at chunk top; base-2 online softmax (native v_exp).
// ---------------------------------------------------------------------------
__global__ __launch_bounds__(128) void attn_kernel(
    const _Float16* __restrict__ q,   // [B*NH, N, DH] (pre-scaled by QSCALE)
    const _Float16* __restrict__ kk,  // [B*NH, N, DH]
    const _Float16* __restrict__ vv,  // [B*NH, DH, N]
    _Float16* __restrict__ outp) {    // [B, N, C]
  __shared__ _Float16 pl[4][16 * 32];
  const int lane = threadIdx.x & 31, wave = threadIdx.x >> 5;
  const int itile = blockIdx.x * 4 + wave;     // 0..255
  const int head = blockIdx.y, b = blockIdx.z;
  const size_t bh = (size_t)(b * NHD + head);
  const int i0 = itile * 16;
  const _Float16* qb = q + bh * NPIX * DH;
  const _Float16* kb = kk + bh * NPIX * DH;
  const _Float16* vb = vv + bh * DH * NPIX;
  _Float16* pw = pl[wave];

  const v16h a0 = load_a(qb, i0, DH, 0, lane);
  const v16h a1 = load_a(qb, i0, DH, 32, lane);

  v8f o[4] = {};
  float mrow[8], lrow[8];
#pragma unroll
  for (int v = 0; v < 8; ++v) { mrow[v] = -1e30f; lrow[v] = 0.f; }

  const int col = lane & 15;
  const int hi8 = (lane >> 4) << 3;

#pragma unroll 2
  for (int jc = 0; jc < NPIX; jc += 32) {
    // ---- issue all global fragment loads for this chunk up front ----
    v16h bk0 = load_b(kb, jc, DH, 0, lane);
    v16h bk1 = load_b(kb, jc, DH, 32, lane);
    v16h bk2 = load_b(kb, jc + 16, DH, 0, lane);
    v16h bk3 = load_b(kb, jc + 16, DH, 32, lane);
    v16h bv[4];
#pragma unroll
    for (int t = 0; t < 4; ++t) bv[t] = load_b(vb, t * 16, NPIX, jc, lane);

    // ---- scores: S[16 x 32] = q_tile @ k_chunk^T (K = dh = 64) ----
    v8f s0 = {}, s1 = {};
    s0 = wmma_f16(a0, bk0, s0);
    s0 = wmma_f16(a1, bk1, s0);
    s1 = wmma_f16(a0, bk2, s1);
    s1 = wmma_f16(a1, bk3, s1);

    // ---- base-2 online softmax (rows live across 16-lane halves) ----
#pragma unroll
    for (int v = 0; v < 8; ++v) {
      float tm = fmaxf(s0[v], s1[v]);
#pragma unroll
      for (int mask = 1; mask < 16; mask <<= 1)
        tm = fmaxf(tm, __shfl_xor(tm, mask, 32));
      const float nm = fmaxf(mrow[v], tm);
      const float corr = exp2f(mrow[v] - nm);
      mrow[v] = nm;
      const float p0 = exp2f(s0[v] - nm);
      const float p1 = exp2f(s1[v] - nm);
      s0[v] = p0;
      s1[v] = p1;
      float rs = p0 + p1;
#pragma unroll
      for (int mask = 1; mask < 16; mask <<= 1)
        rs += __shfl_xor(rs, mask, 32);
      lrow[v] = lrow[v] * corr + rs;
#pragma unroll
      for (int t = 0; t < 4; ++t) o[t][v] *= corr;
    }

    // ---- transpose P (C/D layout -> A layout) through LDS ----
#pragma unroll
    for (int v = 0; v < 8; ++v) {
      pw[(v + hi8) * 32 + col]      = (_Float16)s0[v];
      pw[(v + hi8) * 32 + col + 16] = (_Float16)s1[v];
    }
    const v16h pa = load_a(pw, 0, 32, 0, lane);

    // ---- O[16 x 64] += P @ V^T (K = 32 keys) ----
#pragma unroll
    for (int t = 0; t < 4; ++t) o[t] = wmma_f16(pa, bv[t], o[t]);
  }

  // ---- normalize and store [b, i, head*64 + d] as f16 ----
#pragma unroll
  for (int v = 0; v < 8; ++v) {
    const float inv = 1.0f / lrow[v];
    const int i = i0 + v + hi8;
#pragma unroll
    for (int t = 0; t < 4; ++t) {
      const int d = t * 16 + col;
      outp[((size_t)b * NPIX + i) * CCH + head * DH + d] =
          (_Float16)(o[t][v] * inv);
    }
  }
}

// ---------------------------------------------------------------------------
// Kernel 4: proj GEMM + bias + residual.  32x64 wave tiles, K fully unrolled.
// ---------------------------------------------------------------------------
__global__ __launch_bounds__(128) void proj_gemm_kernel(
    const _Float16* __restrict__ w,    // [256, 256]
    const _Float16* __restrict__ at,   // [B, N, C]
    const float* __restrict__ bias,    // [256]
    const float* __restrict__ x,       // [B, C, N]
    float* __restrict__ out) {         // [B, C, N]
  const int lane = threadIdx.x & 31, wave = threadIdx.x >> 5;
  const int b = blockIdx.z;
  const int n0 = blockIdx.x * 64;
  const int m0 = blockIdx.y * 128 + wave * 32;
  const _Float16* bt = at + (size_t)b * NPIX * CCH;

  v8f acc0[4] = {}, acc1[4] = {};
#pragma unroll
  for (int k = 0; k < CCH; k += 32) {
    v16h a0 = load_a(w, m0, CCH, k, lane);
    v16h a1 = load_a(w, m0 + 16, CCH, k, lane);
    v16h bf[4];
#pragma unroll
    for (int t = 0; t < 4; ++t) bf[t] = load_b(bt, n0 + t * 16, CCH, k, lane);
#pragma unroll
    for (int t = 0; t < 4; ++t) {
      acc0[t] = wmma_f16(a0, bf[t], acc0[t]);
      acc1[t] = wmma_f16(a1, bf[t], acc1[t]);
    }
  }

  const int col = lane & 15;
  const int hi8 = (lane >> 4) << 3;
#pragma unroll
  for (int ms = 0; ms < 2; ++ms) {
    const int m0s = m0 + ms * 16;
    v8f* acc = ms ? acc1 : acc0;
#pragma unroll
    for (int t = 0; t < 4; ++t) {
      const int n = n0 + t * 16 + col;
#pragma unroll
      for (int v = 0; v < 8; ++v) {
        const int o = m0s + v + hi8;
        const size_t idx = ((size_t)b * CCH + o) * NPIX + n;
        out[idx] = acc[t][v] + bias[o] + x[idx];
      }
    }
  }
}

// ---------------------------------------------------------------------------
// Host-side launch
// ---------------------------------------------------------------------------
extern "C" void kernel_launch(void* const* d_in, const int* in_sizes, int n_in,
                              void* d_out, int out_size, void* d_ws,
                              size_t ws_size, hipStream_t stream) {
  const float* x      = (const float*)d_in[0];
  const float* gn_w   = (const float*)d_in[1];
  const float* gn_b   = (const float*)d_in[2];
  const float* w_qkv  = (const float*)d_in[3];
  const float* b_qkv  = (const float*)d_in[4];
  const float* w_proj = (const float*)d_in[5];
  const float* b_proj = (const float*)d_in[6];
  float* out = (float*)d_out;

  // Workspace carve-up (f16 buffers), ~40.5 MB total.
  char* ws = (char*)d_ws;
  _Float16* wqkv16  = (_Float16*)ws; ws += (size_t)3 * CCH * CCH * 2;
  _Float16* wproj16 = (_Float16*)ws; ws += (size_t)CCH * CCH * 2;
  _Float16* h16     = (_Float16*)ws; ws += (size_t)NB * NPIX * CCH * 2;
  _Float16* q16     = (_Float16*)ws; ws += (size_t)NB * NHD * NPIX * DH * 2;
  _Float16* k16     = (_Float16*)ws; ws += (size_t)NB * NHD * NPIX * DH * 2;
  _Float16* v16     = (_Float16*)ws; ws += (size_t)NB * NHD * DH * NPIX * 2;
  _Float16* at16    = (_Float16*)ws; ws += (size_t)NB * NPIX * CCH * 2;

  const int nwq = 3 * CCH * CCH;   // 196608
  const int nwp = CCH * CCH;       // 65536
  cvt_f32_f16<<<(nwq + 255) / 256, 256, 0, stream>>>(w_qkv, wqkv16, nwq);
  cvt_f32_f16<<<(nwp + 255) / 256, 256, 0, stream>>>(w_proj, wproj16, nwp);

  groupnorm_kernel<<<NB * 8, 256, 0, stream>>>(x, gn_w, gn_b, h16);

  qkv_gemm_kernel<<<dim3(NPIX / 64, (3 * CCH) / 128, NB), 128, 0, stream>>>(
      wqkv16, h16, b_qkv, q16, k16, v16);

  attn_kernel<<<dim3(NPIX / 16 / 4, NHD, NB), 128, 0, stream>>>(q16, k16, v16,
                                                                at16);

  proj_gemm_kernel<<<dim3(NPIX / 64, CCH / 128, NB), 128, 0, stream>>>(
      wproj16, at16, b_proj, x, out);
}
